// EMA3D_85899346163
// MI455X (gfx1250) — matrix operand order repaired and checked by reading, not compiled, not run on hardware
//
#include <hip/hip_runtime.h>
#include <hip/hip_bf16.h>
#include <math.h>

// Problem geometry (fixed by reference)
#define BG   16          // b * GROUP
#define CG   8           // channels per group
#define DD   64
#define HH   64
#define WW   64
#define SS   (DD*HH*WW)  // 262144 voxels per (bg, c)
#define PROF 192         // d + h + w
#define EPSN 1e-5f

typedef __attribute__((ext_vector_type(16))) __bf16 v16bf;
typedef __attribute__((ext_vector_type(8)))  __bf16 v8bf;
typedef __attribute__((ext_vector_type(8)))  float  v8f;

// Workspace layout (float offsets)
#define OFF_CAT     0                       // BG*CG*PROF = 24576
#define OFF_ATT     24576                   // BG*CG*PROF
#define OFF_GMEAN   49152                   // BG*CG
#define OFF_GRSTD   49280                   // BG*CG
#define OFF_BNSUM   49408                   // 8 sum + 8 sumsq (atomics)
#define OFF_BNSC    49424                   // 8 scale
#define OFF_BNSH    49432                   // 8 shift
#define OFF_X11     49440                   // 8 (same for all bg)
#define OFF_X2MEAN  49448                   // BG*CG
#define OFF_X21     49576                   // BG*CG
#define OFF_X2PRE   50176                   // BG*CG*SS floats (128 MiB)

// conv kernel dynamic LDS partition (bytes)
#define BSTR        232                     // padded K-stride of im2col panel
#define SM_BLDS     0                       // 256*232 bf16 = 118784 B
#define SM_ALDS     118784                  // 3584 bf16   = 7168 B
#define SM_ILDS     125952                  // 9792 bf16   = 19584 B (9504 + zero pad)
#define SM_LOFF     145536                  // 224 int     = 896 B
#define SM_TOTAL    146432
#define ILDS_PAD    9504                    // invalid-K target (zeroed region)

// ---------------- init: zero the BN accumulators ----------------
__global__ void k_init(float* p) {
    if (threadIdx.x < 16) p[threadIdx.x] = 0.f;
}

// ---------------- directional pools -> cat (bg*cg, 192) ----------------
// thread t owns x-quad (t&15) and y-residue (t>>4); register accumulation,
// one LDS atomic per 16..256 loads instead of 3 per load.
__global__ void k_pool(const float* __restrict__ x, float* __restrict__ cat) {
    __shared__ float pd[DD], ph[HH], pw[WW];
    const int bc = blockIdx.x, tid = threadIdx.x;
    if (tid < 64) { pd[tid] = 0.f; ph[tid] = 0.f; pw[tid] = 0.f; }
    __syncthreads();
    const float4* p4 = (const float4*)(x + (size_t)bc * SS);
    const int xq = tid & 15;   // x = 4*xq + j
    const int yr = tid >> 4;   // y = yr + 16*k
    float xacc[4] = {0.f, 0.f, 0.f, 0.f};
    float yacc[4] = {0.f, 0.f, 0.f, 0.f};
    for (int z = 0; z < DD; ++z) {
        float zacc = 0.f;
#pragma unroll
        for (int k = 0; k < 4; ++k) {
            float4 v = p4[(z << 10) + ((yr + (k << 4)) << 4) + xq];
            float s4 = v.x + v.y + v.z + v.w;
            zacc += s4; yacc[k] += s4;
            xacc[0] += v.x; xacc[1] += v.y; xacc[2] += v.z; xacc[3] += v.w;
        }
#pragma unroll
        for (int o = 16; o; o >>= 1) zacc += __shfl_xor(zacc, o, 32);
        if ((tid & 31) == 0) atomicAdd(&pd[z], zacc);
    }
#pragma unroll
    for (int k = 0; k < 4; ++k) atomicAdd(&ph[yr + (k << 4)], yacc[k]);
#pragma unroll
    for (int j = 0; j < 4; ++j) atomicAdd(&pw[(xq << 2) + j], xacc[j]);
    __syncthreads();
    if (tid < PROF) {
        float v = (tid < 64) ? pd[tid] : (tid < 128 ? ph[tid - 64] : pw[tid - 128]);
        cat[(size_t)bc * PROF + tid] = v * (1.0f / 4096.0f);
    }
}

// ---------------- tiny attention MLP: 1x1x1 conv + BN + ReLU + sigmoid ----------------
__global__ void k_att(const float* __restrict__ cat,
                      const float* __restrict__ w1, const float* __restrict__ b1,
                      const float* __restrict__ g1, const float* __restrict__ be1,
                      float* __restrict__ att) {
    __shared__ float s_sum[CG], s_sq[CG], s_scale[CG], s_shift[CG], s_w[CG * CG];
    const int tid = threadIdx.x;
    if (tid < CG)      { s_sum[tid] = 0.f; s_sq[tid] = 0.f; }
    if (tid < CG * CG) s_w[tid] = w1[tid];
    __syncthreads();
    const int TOT = BG * CG * PROF;
    for (int idx = tid; idx < TOT; idx += 256) {
        int b = idx / (CG * PROF);
        int r = idx % (CG * PROF);
        int o = r / PROF, s = r % PROF;
        float acc = b1[o];
#pragma unroll
        for (int i = 0; i < CG; ++i)
            acc += s_w[o * CG + i] * cat[(size_t)(b * CG + i) * PROF + s];
        att[idx] = acc;
        atomicAdd(&s_sum[o], acc);
        atomicAdd(&s_sq[o], acc * acc);
    }
    __syncthreads();
    if (tid < CG) {
        float n = (float)(BG * PROF);
        float m = s_sum[tid] / n;
        float v = s_sq[tid] / n - m * m;
        float sc = g1[tid] * rsqrtf(v + EPSN);
        s_scale[tid] = sc;
        s_shift[tid] = be1[tid] - m * sc;
    }
    __syncthreads();
    for (int idx = tid; idx < TOT; idx += 256) {
        int o = (idx % (CG * PROF)) / PROF;
        float a = fmaxf(att[idx] * s_scale[o] + s_shift[o], 0.f);
        att[idx] = 1.f / (1.f + __expf(-a));   // sigmoid
    }
}

// ---------------- GroupNorm stats of gx*attention (x1 never materialized) ----------------
__global__ void k_gnstat(const float* __restrict__ x, const float* __restrict__ att,
                         float* __restrict__ gmean, float* __restrict__ grstd) {
    __shared__ float a[PROF];
    __shared__ float rs[256], rq[256];
    const int bc = blockIdx.x, tid = threadIdx.x;
    if (tid < PROF) a[tid] = att[(size_t)bc * PROF + tid];
    __syncthreads();
    const float4* p4 = (const float4*)(x + (size_t)bc * SS);
    const int xq = tid & 15, yr = tid >> 4;
    const float ax0 = a[128 + (xq << 2)], ax1 = a[129 + (xq << 2)];
    const float ax2 = a[130 + (xq << 2)], ax3 = a[131 + (xq << 2)];
    float s = 0.f, q = 0.f;
    for (int z = 0; z < DD; ++z) {
        float az = a[z];
#pragma unroll
        for (int k = 0; k < 4; ++k) {
            int y = yr + (k << 4);
            float ay = az * a[64 + y];
            float4 v = p4[(z << 10) + (y << 4) + xq];
            float f0 = v.x * ay * ax0, f1 = v.y * ay * ax1;
            float f2 = v.z * ay * ax2, f3 = v.w * ay * ax3;
            s += f0 + f1 + f2 + f3;
            q += f0 * f0 + f1 * f1 + f2 * f2 + f3 * f3;
        }
    }
    rs[tid] = s; rq[tid] = q; __syncthreads();
    for (int off = 128; off > 0; off >>= 1) {
        if (tid < off) { rs[tid] += rs[tid + off]; rq[tid] += rq[tid + off]; }
        __syncthreads();
    }
    if (tid == 0) {
        float m = rs[0] / (float)SS;
        float v = rq[0] / (float)SS - m * m;
        gmean[bc] = m;
        grstd[bc] = rsqrtf(v + EPSN);
    }
}

// ---------------- 3x3x3 conv as implicit GEMM on WMMA (bf16 in, f32 acc) ----------------
// M=16 (8 oc zero-padded), K=216 padded to 224 (7 steps of 32), N=16 x-positions.
// Both operands pre-packed in LDS in exact fragment order -> ds_load_b128 pairs.
__global__ void __launch_bounds__(256) k_conv(const float* __restrict__ x,
                                              const float* __restrict__ w3,
                                              const float* __restrict__ b3,
                                              float* __restrict__ x2pre,
                                              float* __restrict__ bnacc) {
    extern __shared__ __align__(16) char smem[];
    __bf16* blds = (__bf16*)(smem + SM_BLDS);  // [col 256][K BSTR]
    __bf16* alds = (__bf16*)(smem + SM_ALDS);  // [m 16][kk 7][hi 2][e 16]
    __bf16* ilds = (__bf16*)(smem + SM_ILDS);  // halo [(ic*3+zz)*6+yy][66] + zero pad
    int*    loff = (int*)   (smem + SM_LOFF);  // K -> halo offset (invalid -> pad)
    __shared__ float bsum[CG], bsq[CG];

    const int tid = threadIdx.x;
    const int bg  = blockIdx.x >> 10;
    const int rem = blockIdx.x & 1023;
    const int z   = rem >> 4;
    const int yb  = (rem & 15) << 2;           // 4 y-rows per block

    if (tid < CG) { bsum[tid] = 0.f; bsq[tid] = 0.f; }
    // K -> halo-offset table (K order: ic*27 + kz*9 + ky*3 + kx; contiguous in x)
    for (int k = tid; k < 224; k += 256) {
        int lo = ILDS_PAD;
        if (k < 216) {
            int ic = k / 27, t = k % 27;
            int kz = t / 9, ky = (t % 9) / 3, kx = t % 3;
            lo = ((ic * 3 + kz) * 6 + ky) * 66 + kx;
        }
        loff[k] = lo;
    }
    // A fragments pre-packed in contiguous k order (k = 16*hi + e), same as B
    for (int j = tid; j < 3584; j += 256) {
        int e = j & 15, hi = (j >> 4) & 1, kk = (j >> 5) % 7, m = (j >> 5) / 7;
        int hv = e >> 1;
        int kl = e + (hi ? 16 : 0);
        int K  = kk * 32 + kl;
        float wv = (m < CG && K < 216) ? w3[m * 216 + K] : 0.f;
        alds[j] = (__bf16)wv;
    }
    // halo tile (coalesced global reads) + zeroed pad region for invalid K
    for (int j = tid; j < 9792; j += 256) {
        float v = 0.f;
        if (j < 9504) {
            int ic = j / 1188, r = j % 1188;
            int zz = r / 396,  r2 = r % 396;
            int yy = r2 / 66,  xx = r2 % 66;
            int gz = z + zz - 1, gy = yb + yy - 1, gxp = xx - 1;
            if ((unsigned)gz < 64u && (unsigned)gy < 64u && (unsigned)gxp < 64u)
                v = x[(size_t)(bg * CG + ic) * SS + (gz << 12) + (gy << 6) + gxp];
        }
        ilds[j] = (__bf16)v;
    }
    __syncthreads();
    // im2col panel: one column per thread; loff[K] broadcast across lanes, branch-free
    {
        const int add = (tid >> 6) * 66 + (tid & 63);   // yl*66 + xg
        __bf16* dst = blds + tid * BSTR;
        for (int K = 0; K < 224; ++K)
            dst[K] = ilds[loff[K] + add];
    }
    __syncthreads();

    float bias[CG];
#pragma unroll
    for (int r = 0; r < CG; ++r) bias[r] = b3[r];

    const int wv  = tid >> 5, lane = tid & 31;
    const int xt  = wv & 3;                    // x-tile
    const int ylb = wv >> 2;                   // yl base (0..1), +2 on 2nd pass
    const int m   = lane & 15, hi = lane >> 4;
    const __bf16* arow = alds + m * 224 + hi * 16;

#pragma unroll
    for (int t = 0; t < 2; ++t) {
        const int yl = ylb + (t << 1);
        const __bf16* bcol = blds + (size_t)(((yl << 6) + (xt << 4) + m) * BSTR) + hi * 16;
        v8f c = {};
#pragma unroll
        for (int kk = 0; kk < 7; ++kk) {
            union { v16bf v; v8bf h[2]; } au, bu;
            const v8bf* ap = (const v8bf*)(arow + kk * 32);
            const v8bf* bp = (const v8bf*)(bcol + kk * 32);
            au.h[0] = ap[0]; au.h[1] = ap[1];
            bu.h[0] = bp[0]; bu.h[1] = bp[1];
            c = __builtin_amdgcn_wmma_f32_16x16x32_bf16(
                    false, au.v, false, bu.v, (short)0, c, false, false);
        }
        if (!hi) {   // D rows 0..7 (lanes 0-15) are the 8 real output channels
            size_t base = (size_t)bg * CG * SS + ((size_t)z << 12) +
                          (size_t)((yb + yl) << 6) + (size_t)((xt << 4) + m);
#pragma unroll
            for (int r = 0; r < CG; ++r) {
                float val = c[r] + bias[r];
                x2pre[base + (size_t)r * SS] = val;
                atomicAdd(&bsum[r], val);
                atomicAdd(&bsq[r],  val * val);
            }
        }
    }
    __syncthreads();
    if (tid < CG) {
        atomicAdd(&bnacc[tid],      bsum[tid]);
        atomicAdd(&bnacc[CG + tid], bsq[tid]);
    }
}

// ---------------- BN3 finalize + x11 = softmax(gn_b) (mean of GN output == beta) ----------------
__global__ void k_bnfin(const float* __restrict__ bnacc,
                        const float* __restrict__ g3, const float* __restrict__ be3,
                        const float* __restrict__ gnb,
                        float* __restrict__ scale, float* __restrict__ shift,
                        float* __restrict__ x11) {
    int tid = threadIdx.x;
    if (tid < CG) {
        float n = (float)BG * (float)SS;
        float m = bnacc[tid] / n;
        float v = bnacc[CG + tid] / n - m * m;
        float sc = g3[tid] * rsqrtf(v + EPSN);
        scale[tid] = sc;
        shift[tid] = be3[tid] - m * sc;
    }
    if (tid == 0) {
        float mx = gnb[0];
        for (int i = 1; i < CG; ++i) mx = fmaxf(mx, gnb[i]);
        float e[CG], s = 0.f;
        for (int i = 0; i < CG; ++i) { e[i] = __expf(gnb[i] - mx); s += e[i]; }
        for (int i = 0; i < CG; ++i) x11[i] = e[i] / s;
    }
}

// ---------------- per-(bg,c) spatial mean of relu(bn(x2pre)) ----------------
__global__ void k_x2mean(const float* __restrict__ x2pre,
                         const float* __restrict__ scale, const float* __restrict__ shift,
                         float* __restrict__ x2mean) {
    const int bc = blockIdx.x, c = bc & 7, tid = threadIdx.x;
    const float4* p4 = (const float4*)(x2pre + (size_t)bc * SS);
    float sc = scale[c], sh = shift[c], s = 0.f;
    for (int i = tid; i < SS / 4; i += 256) {
        __builtin_prefetch(p4 + i + 256 * 4, 0, 0);
        float4 v = p4[i];
        s += fmaxf(v.x * sc + sh, 0.f) + fmaxf(v.y * sc + sh, 0.f) +
             fmaxf(v.z * sc + sh, 0.f) + fmaxf(v.w * sc + sh, 0.f);
    }
    __shared__ float rs[256];
    rs[tid] = s; __syncthreads();
    for (int off = 128; off > 0; off >>= 1) {
        if (tid < off) rs[tid] += rs[tid + off];
        __syncthreads();
    }
    if (tid == 0) x2mean[bc] = rs[0] / (float)SS;
}

// ---------------- x21 = softmax over channels ----------------
__global__ void k_soft(const float* __restrict__ x2mean, float* __restrict__ x21) {
    int b = threadIdx.x;
    if (b >= BG) return;
    const float* m = x2mean + b * CG;
    float mx = m[0];
    for (int i = 1; i < CG; ++i) mx = fmaxf(mx, m[i]);
    float e[CG], s = 0.f;
    for (int i = 0; i < CG; ++i) { e[i] = __expf(m[i] - mx); s += e[i]; }
    for (int i = 0; i < CG; ++i) x21[b * CG + i] = e[i] / s;
}

// ---------------- fused final: recompute x1, apply BN-ReLU to x2pre, gate ----------------
__global__ void k_final(const float* __restrict__ x, const float* __restrict__ att,
                        const float* __restrict__ x2pre,
                        const float* __restrict__ gmean, const float* __restrict__ grstd,
                        const float* __restrict__ gng, const float* __restrict__ gnb,
                        const float* __restrict__ scale, const float* __restrict__ shift,
                        const float* __restrict__ x11, const float* __restrict__ x21,
                        float* __restrict__ out) {
    __shared__ float s_att[CG * PROF];
    __shared__ float s_gm[CG], s_gr[CG], s_x21[CG], s_sc[CG], s_sh[CG], s_x11[CG], s_gb[CG];
    const int bg = blockIdx.x >> 8;            // 256 blocks per bg (float4 granularity)
    const int tid = threadIdx.x;
    for (int j = tid; j < CG * PROF; j += 256)
        s_att[j] = att[(size_t)bg * CG * PROF + j];
    if (tid < CG) {
        int bi = bg * CG + tid;
        s_gm[tid]  = gmean[bi];
        s_gr[tid]  = grstd[bi] * gng[tid];     // fold GN gamma into rstd
        s_x21[tid] = x21[bi];
        s_sc[tid]  = scale[tid];
        s_sh[tid]  = shift[tid];
        s_x11[tid] = x11[tid];
        s_gb[tid]  = gnb[tid];
    }
    __syncthreads();
    const int n4 = ((blockIdx.x & 255) << 8) + tid;   // float4 index in (bg) volume
    const int n  = n4 << 2;
    const int z = n >> 12, y = (n >> 6) & 63, x0 = n & 63;
    float4 xv[CG];
    float wt0 = 0.f, wt1 = 0.f, wt2 = 0.f, wt3 = 0.f;
#pragma unroll
    for (int c = 0; c < CG; ++c) {
        size_t bi = (size_t)(bg * CG + c);
        float4 v = ((const float4*)x)[bi * (SS / 4) + n4];
        xv[c] = v;
        const float* ap = s_att + c * PROF;
        float ah = ap[z] * ap[64 + y];
        float a0 = ah * ap[128 + x0], a1 = ah * ap[129 + x0];
        float a2 = ah * ap[130 + x0], a3 = ah * ap[131 + x0];
        float gm = s_gm[c], gr = s_gr[c], gb = s_gb[c];
        float w21 = s_x21[c], w11 = s_x11[c], sc = s_sc[c], sh = s_sh[c];
        float4 t = ((const float4*)x2pre)[bi * (SS / 4) + n4];
        wt0 += w11 * fmaxf(t.x * sc + sh, 0.f) + w21 * ((v.x * a0 - gm) * gr + gb);
        wt1 += w11 * fmaxf(t.y * sc + sh, 0.f) + w21 * ((v.y * a1 - gm) * gr + gb);
        wt2 += w11 * fmaxf(t.z * sc + sh, 0.f) + w21 * ((v.z * a2 - gm) * gr + gb);
        wt3 += w11 * fmaxf(t.w * sc + sh, 0.f) + w21 * ((v.w * a3 - gm) * gr + gb);
    }
    float sg0 = 1.f / (1.f + __expf(-wt0));
    float sg1 = 1.f / (1.f + __expf(-wt1));
    float sg2 = 1.f / (1.f + __expf(-wt2));
    float sg3 = 1.f / (1.f + __expf(-wt3));
#pragma unroll
    for (int c = 0; c < CG; ++c) {
        float4 o;
        o.x = xv[c].x * sg0; o.y = xv[c].y * sg1;
        o.z = xv[c].z * sg2; o.w = xv[c].w * sg3;
        ((float4*)out)[(size_t)(bg * CG + c) * (SS / 4) + n4] = o;
    }
}

extern "C" void kernel_launch(void* const* d_in, const int* in_sizes, int n_in,
                              void* d_out, int out_size, void* d_ws, size_t ws_size,
                              hipStream_t stream) {
    const float* x   = (const float*)d_in[0];
    const float* w1  = (const float*)d_in[1];
    const float* b1  = (const float*)d_in[2];
    const float* g1  = (const float*)d_in[3];
    const float* be1 = (const float*)d_in[4];
    const float* w3  = (const float*)d_in[5];
    const float* b3  = (const float*)d_in[6];
    const float* g3  = (const float*)d_in[7];
    const float* be3 = (const float*)d_in[8];
    const float* gng = (const float*)d_in[9];
    const float* gnb = (const float*)d_in[10];
    float* out = (float*)d_out;
    float* ws  = (float*)d_ws;

    float* cat    = ws + OFF_CAT;
    float* att    = ws + OFF_ATT;
    float* gmean  = ws + OFF_GMEAN;
    float* grstd  = ws + OFF_GRSTD;
    float* bnacc  = ws + OFF_BNSUM;
    float* bnsc   = ws + OFF_BNSC;
    float* bnsh   = ws + OFF_BNSH;
    float* x11    = ws + OFF_X11;
    float* x2mean = ws + OFF_X2MEAN;
    float* x21    = ws + OFF_X21;
    float* x2pre  = ws + OFF_X2PRE;

    (void)hipFuncSetAttribute((const void*)k_conv,
                              hipFuncAttributeMaxDynamicSharedMemorySize, SM_TOTAL);

    k_init  <<<1, 32, 0, stream>>>(bnacc);
    k_pool  <<<BG * CG, 256, 0, stream>>>(x, cat);
    k_att   <<<1, 256, 0, stream>>>(cat, w1, b1, g1, be1, att);
    k_gnstat<<<BG * CG, 256, 0, stream>>>(x, att, gmean, grstd);
    k_conv  <<<BG * DD * (HH / 4), 256, SM_TOTAL, stream>>>(x, w3, b3, x2pre, bnacc);
    k_bnfin <<<1, 32, 0, stream>>>(bnacc, g3, be3, gnb, bnsc, bnsh, x11);
    k_x2mean<<<BG * CG, 256, 0, stream>>>(x2pre, bnsc, bnsh, x2mean);
    k_soft  <<<1, 32, 0, stream>>>(x2mean, x21);
    k_final <<<BG * (SS / 1024), 256, 0, stream>>>(x, att, x2pre, gmean, grstd,
                                                   gng, gnb, bnsc, bnsh, x11, x21, out);
    (void)in_sizes; (void)n_in; (void)out_size; (void)ws_size;
}
